// ProtoTSNet_34351148434061
// MI455X (gfx1250) — matrix operand order repaired and checked by reading, not compiled, not run on hardware
//
#include <hip/hip_runtime.h>
#include <hip/hip_bf16.h>
#include <math.h>

// ---- problem constants -----------------------------------------------------
#define BATCH   32
#define FIN     16
#define LSEQ    8192
#define LATENT  64
#define LF      8190      // length after K=3 valid conv
#define LFP     8192      // padded time stride of the f16 feature buffer
#define PN      200       // prototypes
#define PLEN    16
#define KDIM    1024      // LATENT*PLEN, contraction size of the xp GEMM
#define LO      8175      // LF - PLEN + 1
#define NCLS    10
#define EPSV    1e-4f

#define TT      128       // time positions per k2 block (8 N-tiles of 16)
#define MT      13        // ceil(PN/16) proto M-tiles

typedef __attribute__((ext_vector_type(16))) _Float16 v16h;
typedef __attribute__((ext_vector_type(8)))  _Float16 v8h;
typedef __attribute__((ext_vector_type(8)))  float    v8f;

// ---- workspace layout (bytes) ---------------------------------------------
static const size_t OFF_P16 = 0;                                   // 200*1024*2 = 409600
static const size_t OFF_P2  = 409600;                              // 200*4 = 800
static const size_t OFF_F   = 410624;                              // 32*64*8192*2 = 33554432
static const size_t OFF_G   = OFF_F + (size_t)33554432;            // 32*8190*4 = 1048320

// ============================================================================
// k0: protos fp32 -> f16 (flat K order = c*16+k, same as input layout) + p2
// ============================================================================
__global__ __launch_bounds__(256) void k0_prep(const float* __restrict__ protos,
                                               _Float16* __restrict__ p16,
                                               float* __restrict__ p2) {
    int p = blockIdx.x, tid = threadIdx.x;
    __shared__ float red[256];
    float s = 0.f;
    for (int i = tid; i < KDIM; i += 256) {
        float v = protos[(size_t)p * KDIM + i];
        p16[(size_t)p * KDIM + i] = (_Float16)v;
        s += v * v;
    }
    red[tid] = s; __syncthreads();
    for (int o = 128; o > 0; o >>= 1) {
        if (tid < o) red[tid] += red[tid + o];
        __syncthreads();
    }
    if (tid == 0) p2[p] = red[0];
}

// ============================================================================
// k_init: min-distance accumulators <- +inf bit pattern (uint ordering ==
// float ordering for non-negative values)
// ============================================================================
__global__ __launch_bounds__(256) void k_init(unsigned* __restrict__ minbits) {
    int i = blockIdx.x * 256 + threadIdx.x;
    if (i < BATCH * PN) minbits[i] = 0x7F800000u;   // +inf
}

// ============================================================================
// k1: fused enc-conv(16->64,K=3)+ReLU + 1x1 conv(64->64)+ReLU.
// Writes f16 features f[b][c][t] (stride LFP, tail zero-padded) and
// g[b][t] = sum_c f^2.
// ============================================================================
__global__ __launch_bounds__(256) void k1_feat(const float* __restrict__ x,
                                               const float* __restrict__ ew,
                                               const float* __restrict__ aw,
                                               _Float16* __restrict__ f16,
                                               float* __restrict__ g) {
    __shared__ float xs[FIN][67];        // input slab, 64 outputs need t..t+2
    __shared__ float ews[LATENT * 48];   // enc weights [co][ci*3+k]
    __shared__ float aws[LATENT * 64];   // addon weights [co][ci]
    __shared__ float es[LATENT][65];     // enc activations
    __shared__ float gp[4][64];          // per-group partial sum of squares

    int b = blockIdx.y, t0 = blockIdx.x * 64, tid = threadIdx.x;

    for (int i = tid; i < LATENT * 48; i += 256) ews[i] = ew[i];
    for (int i = tid; i < LATENT * 64; i += 256) aws[i] = aw[i];
    const float* xb = x + (size_t)b * FIN * LSEQ;
    for (int i = tid; i < FIN * 66; i += 256) {
        int ci = i / 66, c = i % 66, gt = t0 + c;
        xs[ci][c] = (gt < LSEQ) ? xb[(size_t)ci * LSEQ + gt] : 0.f;
    }
    __syncthreads();

    int t = tid & 63, grp = tid >> 6;          // 4 groups x 16 channels
    // stage A: encoder conv
    for (int i = 0; i < 16; i++) {
        int co = grp * 16 + i;
        float acc = 0.f;
        for (int ci = 0; ci < FIN; ci++)
            for (int k = 0; k < 3; k++)
                acc += xs[ci][t + k] * ews[co * 48 + ci * 3 + k];
        es[co][t] = acc > 0.f ? acc : 0.f;
    }
    __syncthreads();

    // stage B: 1x1 conv + ReLU, emit f16 + squared sum
    int gt = t0 + t;                           // gt <= 8191 < LFP always
    _Float16* fb = f16 + (size_t)b * LATENT * LFP;
    float h2 = 0.f;
    for (int i = 0; i < 16; i++) {
        int co = grp * 16 + i;
        float acc = 0.f;
        for (int ci = 0; ci < LATENT; ci++) acc += aws[co * 64 + ci] * es[ci][t];
        acc = acc > 0.f ? acc : 0.f;
        h2 += acc * acc;
        fb[(size_t)co * LFP + gt] = (gt < LF) ? (_Float16)acc : (_Float16)0.f;
    }
    gp[grp][t] = h2;
    __syncthreads();
    if (tid < 64 && gt < LF)
        g[(size_t)b * LF + gt] = gp[0][t] + gp[1][t] + gp[2][t] + gp[3][t];
}

// ============================================================================
// k2: the WMMA kernel. Per block: one batch b, TT=128 time positions.
// 8 waves; wave w owns N-tile (16 t's). Loop order: K-chunk outer, 13 proto
// M-tiles inner with 13 live f32 accumulators, so per K-chunk the B (patch)
// fragment is loaded once from LDS and 13 independent v_wmma ops issue
// back-to-back (no D->A/B RAW hazard between different accumulators) while
// A fragments stream from the L2-resident f16 protos.
//   A (protos 16x32 f16): lane L row M=L%16 holds K=Kb..Kb+15,
//     Kb = 16*(L>=16) -> two aligned 16B global loads.
//   B (patch 32x16 f16): lane n<16 holds col n K=0..15; lane n+16 K=16..31.
//     B[k][n] = f[b][2*kc+hs][t0 + nt*16 + n + e] -> 16 contiguous LDS halves.
//   C/D (16x16 f32): vgpr r, lanes 0-15 -> M=r, lanes 16-31 -> M=r+8.
// ============================================================================
__global__ __launch_bounds__(256) void k2_dist(const _Float16* __restrict__ f16,
                                               const _Float16* __restrict__ p16,
                                               const float* __restrict__ p2,
                                               const float* __restrict__ g,
                                               unsigned* __restrict__ minbits) {
    __shared__ _Float16 fs[LATENT][TT + 16];   // 144-half rows (288B, 16B-mult)
    __shared__ float    gs[TT + 16];           // sum-of-squares slab

    int b = blockIdx.y;
    int t0 = blockIdx.x * TT;
    int tid = threadIdx.x;

    // ---- stage feature slab: aligned v8h chunks (18 chunks of 8 per row) ---
    const _Float16* fb = f16 + (size_t)b * LATENT * LFP;
    for (int i = tid; i < LATENT * 18; i += 256) {
        int row = i / 18, c8 = i % 18;
        int col = c8 * 8, gt = t0 + col;       // gt multiple of 8 -> 16B aligned
        if (gt + 8 <= LFP) {
            *(v8h*)&fs[row][col] = *(const v8h*)(fb + (size_t)row * LFP + gt);
        } else {                                // tail: guard, zero-fill
            for (int e = 0; e < 8; e++)
                fs[row][col + e] = (gt + e < LFP) ? fb[(size_t)row * LFP + gt + e]
                                                  : (_Float16)0.f;
        }
    }
    const float* gb = g + (size_t)b * LF;
    for (int i = tid; i < TT + 15; i += 256) {
        int gt = t0 + i;
        gs[i] = (gt < LF) ? gb[gt] : 0.f;
    }
    __syncthreads();

    int wave = tid >> 5, lane = tid & 31;
    int n = lane & 15, hs = lane >> 4;     // half-wave select
    int nt = wave;                          // N-tile id (0..7)
    int t = t0 + nt * 16 + n;               // this lane's time position

    // sliding-window sum of squares (collapsed x2_patch_sum)
    float w2 = 0.f;
    for (int k = 0; k < 16; k++) w2 += gs[nt * 16 + n + k];
    if (t >= LO) w2 = __builtin_inff();     // mask invalid time positions

    int Kb = hs ? 16 : 0;
    __builtin_prefetch(p16 + (size_t)n * KDIM, 0, 0);  // warm L2 path to protos

    v8f acc[MT];
    #pragma unroll
    for (int i = 0; i < MT; i++) acc[i] = (v8f){};

    for (int kc = 0; kc < 32; kc++) {
        // B fragment: 16 contiguous halves of one LDS row, shared by all MT
        const _Float16* bp = &fs[2 * kc + hs][nt * 16 + n];
        v16h bb;
        #pragma unroll
        for (int e = 0; e < 16; e++) bb[e] = bp[e];

        #pragma unroll
        for (int mt = 0; mt < MT; mt++) {
            int ma = mt * 16 + n; if (ma > PN - 1) ma = PN - 1;  // clamp pad rows
            const v8h* ap = (const v8h*)(p16 + (size_t)ma * KDIM + Kb + kc * 32);
            v8h alo = ap[0];        // K = Kb .. Kb+7
            v8h ahi = ap[1];        // K = Kb+8 .. Kb+15
            v16h a;
            #pragma unroll
            for (int e = 0; e < 8; e++) { a[e] = alo[e]; a[8 + e] = ahi[e]; }
            acc[mt] = __builtin_amdgcn_wmma_f32_16x16x32_f16(
                false, a, false, bb, (short)0, acc[mt], false, false);
        }
    }

    // ---- distance + time-min + global reduction ---------------------------
    #pragma unroll
    for (int mt = 0; mt < MT; mt++) {
        #pragma unroll
        for (int r = 0; r < 8; r++) {
            int m = mt * 16 + r + 8 * hs;
            float pp = p2[m < PN ? m : PN - 1];
            float d = w2 - 2.f * acc[mt][r] + pp;
            d = d < 0.f ? 0.f : d;                       // ReLU
            for (int off = 1; off < 16; off <<= 1) {     // min over 16 t's
                float o = __shfl_xor(d, off, 16);
                d = o < d ? o : d;
            }
            if (n == 0 && m < PN)
                atomicMin(&minbits[b * PN + m], __float_as_uint(d));
        }
    }
}

// ============================================================================
// k3: log similarity + linear head
// ============================================================================
__global__ __launch_bounds__(256) void k3_logits(const float* __restrict__ md,
                                                 const float* __restrict__ lw,
                                                 float* __restrict__ logits) {
    int b = blockIdx.x, tid = threadIdx.x;
    __shared__ float acts[PN];
    if (tid < PN) {
        float m = md[b * PN + tid];
        acts[tid] = logf((m + 1.f) / (m + EPSV));
    }
    __syncthreads();
    if (tid < NCLS) {
        float s = 0.f;
        for (int p = 0; p < PN; p++) s += acts[p] * lw[tid * PN + p];
        logits[b * NCLS + tid] = s;
    }
}

// ============================================================================
extern "C" void kernel_launch(void* const* d_in, const int* in_sizes, int n_in,
                              void* d_out, int out_size, void* d_ws, size_t ws_size,
                              hipStream_t stream) {
    const float* x       = (const float*)d_in[0];
    const float* enc_w   = (const float*)d_in[1];
    const float* addon_w = (const float*)d_in[2];
    const float* protos  = (const float*)d_in[3];
    const float* last_w  = (const float*)d_in[4];

    char* ws = (char*)d_ws;
    _Float16* p16 = (_Float16*)(ws + OFF_P16);
    float*    p2  = (float*)   (ws + OFF_P2);
    _Float16* f16 = (_Float16*)(ws + OFF_F);
    float*    g   = (float*)   (ws + OFF_G);

    float* out        = (float*)d_out;           // [0,320): logits
    float* mind       = out + BATCH * NCLS;      // [320,6720): min_distances
    unsigned* minbits = (unsigned*)mind;

    k0_prep<<<PN, 256, 0, stream>>>(protos, p16, p2);
    k_init <<<(BATCH * PN + 255) / 256, 256, 0, stream>>>(minbits);
    k1_feat<<<dim3((LF + 63) / 64, BATCH), 256, 0, stream>>>(x, enc_w, addon_w, f16, g);
    k2_dist<<<dim3((LO + TT - 1) / TT, BATCH), 256, 0, stream>>>(f16, p16, p2, g, minbits);
    k3_logits<<<BATCH, 256, 0, stream>>>(mind, last_w, out);
}
